// BatchTaskAlignedAssigner_16604343566616
// MI455X (gfx1250) — matrix operand order repaired and checked by reading, not compiled, not run on hardware
//
#include <hip/hip_runtime.h>
#include <hip/hip_bf16.h>
#include <float.h>

#define B_ 32
#define G_ 64
#define P_ 8400
#define C_ 80
#define K_ 13
#define EPS_ 1e-9f
#define IOU_EPS_ 1e-6f

typedef __attribute__((ext_vector_type(16))) _Float16 v16h;
typedef __attribute__((ext_vector_type(8)))  float    v8f;

// ---------------------------------------------------------------------------
// K1: align = score[b,p,label(b,g)] * iou(b,g,p)^6 ; also store iou.
// One block per (b,g) row; threads stride over P (coalesced).
// ---------------------------------------------------------------------------
__global__ void k_align_iou(const float* __restrict__ ps, const float* __restrict__ pb,
                            const int* __restrict__ gl, const float* __restrict__ gb,
                            float* __restrict__ al, float* __restrict__ ov) {
  int bg = blockIdx.x;
  int b  = bg / G_;
  float gx1 = gb[bg * 4 + 0], gy1 = gb[bg * 4 + 1];
  float gx2 = gb[bg * 4 + 2], gy2 = gb[bg * 4 + 3];
  int   lab = gl[bg];
  float area_g = (gx2 - gx1) * (gy2 - gy1);
  size_t rowo = (size_t)bg * P_;
  for (int p = threadIdx.x; p < P_; p += blockDim.x) {
    size_t pbo = ((size_t)b * P_ + p) * 4;
    float x1 = pb[pbo + 0], y1 = pb[pbo + 1], x2 = pb[pbo + 2], y2 = pb[pbo + 3];
    float iw = fmaxf(fminf(gx2, x2) - fmaxf(gx1, x1), 0.f);
    float ih = fmaxf(fminf(gy2, y2) - fmaxf(gy1, y1), 0.f);
    float inter = iw * ih;
    float area_p = (x2 - x1) * (y2 - y1);
    float uni = fmaxf(area_g + area_p - inter, IOU_EPS_);
    float iou = inter / uni;
    const float* sp = ps + ((size_t)b * P_ + p) * C_ + lab;
    __builtin_prefetch(sp + (size_t)blockDim.x * C_, 0, 1);  // global_prefetch_b8
    float s  = *sp;
    float i2 = iou * iou;
    al[rowo + p] = s * i2 * i2 * i2;
    ov[rowo + p] = iou;
  }
}

__global__ void k_zero(float* __restrict__ buf, size_t n) {
  size_t i = (size_t)blockIdx.x * blockDim.x + threadIdx.x;
  if (i < n) buf[i] = 0.f;
}

// ---------------------------------------------------------------------------
// K2: top-13 of align*mask_in_gts per (b,g) row, scatter mask*pad into pos_mask.
// Tie-break = smaller index (matches jax.lax.top_k ordering).
// ---------------------------------------------------------------------------
__global__ void k_topk_scatter(const float* __restrict__ al, const float* __restrict__ pri,
                               const float* __restrict__ gb, const float* __restrict__ pf,
                               float* __restrict__ pos) {
  const int NT = 256;
  int bg = blockIdx.x;
  float pad = pf[bg];
  if (pad <= 0.f) return;  // row stays all-zero (reference: counts>1 -> 0)
  float gx1 = gb[bg * 4 + 0], gy1 = gb[bg * 4 + 1];
  float gx2 = gb[bg * 4 + 2], gy2 = gb[bg * 4 + 3];
  size_t rowo = (size_t)bg * P_;
  int tid = threadIdx.x;

  float lv[K_]; int li[K_];
  #pragma unroll
  for (int k = 0; k < K_; ++k) { lv[k] = -1.f; li[k] = 0x7fffffff; }

  for (int p = tid; p < P_; p += NT) {
    float px = pri[2 * p], py = pri[2 * p + 1];
    float d = fminf(fminf(px - gx1, py - gy1), fminf(gx2 - px, gy2 - py));
    float v = (d > EPS_) ? al[rowo + p] : 0.f;
    if (v > lv[K_ - 1]) {                 // strict >: stable (earlier p wins ties)
      int q = K_ - 1;
      lv[q] = v; li[q] = p;
      while (q > 0 && lv[q] > lv[q - 1]) {
        float tv = lv[q]; lv[q] = lv[q - 1]; lv[q - 1] = tv;
        int   ti = li[q]; li[q] = li[q - 1]; li[q - 1] = ti;
        --q;
      }
    }
  }

  __shared__ float sv[NT];
  __shared__ int   si[NT];
  __shared__ int   sowner[NT];
  __shared__ int   sel[K_];
  int ptr = 0;
  for (int k = 0; k < K_; ++k) {
    sv[tid] = (ptr < K_) ? lv[ptr] : -2.f;
    si[tid] = (ptr < K_) ? li[ptr] : 0x7fffffff;
    sowner[tid] = tid;
    __syncthreads();
    for (int s = NT >> 1; s > 0; s >>= 1) {
      if (tid < s) {
        float va = sv[tid], vb = sv[tid + s];
        int   ia = si[tid], ib = si[tid + s];
        if (vb > va || (vb == va && ib < ia)) {
          sv[tid] = vb; si[tid] = ib; sowner[tid] = sowner[tid + s];
        }
      }
      __syncthreads();
    }
    int winner = sowner[0];
    if (tid == 0) sel[k] = si[0];
    __syncthreads();
    if (tid == winner) ++ptr;
  }
  if (tid < K_) {
    int p = sel[tid];
    float px = pri[2 * p], py = pri[2 * p + 1];
    float d = fminf(fminf(px - gx1, py - gy1), fminf(gx2 - px, gy2 - py));
    pos[rowo + p] = (d > EPS_) ? pad : 0.f;  // is_in_topk(=1) * mask_in_gts * pad
  }
}

// ---------------------------------------------------------------------------
// K3 (WMMA): fg[b,p] = sum_g pos_mask[b,g,p].
// A = ones(16x32) f16, B[k][n] = pos_mask[g = kk*32 + k][p0 + n] (lane<->N,
// slot<->K, k = 16*half + slot), two K-chunks cover G=64. D row0 = column sums.
// One wave per 16-prior tile; counts <= 64 are exact in f16.
// ---------------------------------------------------------------------------
__global__ void k_fg_wmma(const float* __restrict__ pos, float* __restrict__ fgw) {
  const int PT = P_ / 16;                 // 525 tiles per batch
  int tile = blockIdx.x;
  int b  = tile / PT;
  int p0 = (tile % PT) * 16;
  int lane = threadIdx.x;                 // 0..31, single wave32

  v16h a;
  #pragma unroll
  for (int j = 0; j < 16; ++j) a[j] = (_Float16)1.0f;

  v8f c = {};
  #pragma unroll
  for (int kk = 0; kk < 2; ++kk) {
    int g = kk * 32 + (lane >> 4) * 16;   // lane half <-> K block of B
    const float* rowp = pos + ((size_t)b * G_ + g) * P_ + p0 + (lane & 15);
    v16h bm;
    #pragma unroll
    for (int j = 0; j < 16; ++j) bm[j] = (_Float16)rowp[(size_t)j * P_];   // slot <-> K
    c = __builtin_amdgcn_wmma_f32_16x16x32_f16(
        /*neg_a=*/false, a, /*neg_b=*/false, bm,
        /*c_mod=*/(short)0, c, /*reuse_a=*/false, /*reuse_b=*/false);
  }
  // D layout: VGPR0, lanes 0..15 -> M=0, N=lane
  if (lane < 16) fgw[(size_t)b * P_ + p0 + lane] = c[0];
}

// ---------------------------------------------------------------------------
// K4: resolve multi-assigned priors (fg>1 -> one-hot of IoU argmax over G),
// produce assigned gt index and final fg count.
// ---------------------------------------------------------------------------
__global__ void k_multi(float* __restrict__ pos, const float* __restrict__ ov,
                        const float* __restrict__ fgw, int* __restrict__ agb,
                        float* __restrict__ fgf) {
  int i = blockIdx.x * blockDim.x + threadIdx.x;
  if (i >= B_ * P_) return;
  int b = i / P_, p = i % P_;
  size_t base = (size_t)b * G_ * P_ + p;
  int first1 = -1, oarg = 0;
  float omax = -FLT_MAX;
  for (int g = 0; g < G_; ++g) {
    float pm = pos[base + (size_t)g * P_];
    float o  = ov[base + (size_t)g * P_];
    if (first1 < 0 && pm > 0.5f) first1 = g;
    if (o > omax) { omax = o; oarg = g; }   // strict >: first max (argmax)
  }
  float f = fgw[i];                         // exact integer from WMMA f16 path
  if (f > 1.5f) {
    for (int g = 0; g < G_; ++g) pos[base + (size_t)g * P_] = (g == oarg) ? 1.f : 0.f;
    agb[i] = oarg; fgf[i] = 1.f;
  } else {
    agb[i] = (first1 >= 0) ? first1 : 0;
    fgf[i] = f;
  }
}

// ---------------------------------------------------------------------------
// K5: per-(b,g) maxima of align*pos and ovl*pos over P.
// ---------------------------------------------------------------------------
__global__ void k_rowmax(const float* __restrict__ al, const float* __restrict__ ov,
                         const float* __restrict__ pos, float* __restrict__ pa,
                         float* __restrict__ po) {
  const int NT = 256;
  int bg = blockIdx.x, tid = threadIdx.x;
  size_t rowo = (size_t)bg * P_;
  float ma = 0.f, mo = 0.f;                 // all candidates are >= 0
  for (int p = tid; p < P_; p += NT) {
    float pm = pos[rowo + p];
    ma = fmaxf(ma, al[rowo + p] * pm);
    mo = fmaxf(mo, ov[rowo + p] * pm);
  }
  __shared__ float sa[NT], sb[NT];
  sa[tid] = ma; sb[tid] = mo; __syncthreads();
  for (int s = NT >> 1; s > 0; s >>= 1) {
    if (tid < s) {
      sa[tid] = fmaxf(sa[tid], sa[tid + s]);
      sb[tid] = fmaxf(sb[tid], sb[tid + s]);
    }
    __syncthreads();
  }
  if (tid == 0) { pa[bg] = sa[0]; po[bg] = sb[0]; }
}

// ---------------------------------------------------------------------------
// K6: per-(b,p) norm + labels + bboxes + fg outputs.
// ---------------------------------------------------------------------------
__global__ void k_assign(const float* __restrict__ al, const float* __restrict__ pos,
                         const float* __restrict__ pa, const float* __restrict__ po,
                         const int* __restrict__ agb, const float* __restrict__ fgf,
                         const int* __restrict__ gl, const float* __restrict__ gb,
                         float* __restrict__ out_lab, float* __restrict__ out_bb,
                         float* __restrict__ out_fg, int* __restrict__ labbuf,
                         float* __restrict__ normbuf) {
  int i = blockIdx.x * blockDim.x + threadIdx.x;
  if (i >= B_ * P_) return;
  int b = i / P_, p = i % P_;
  size_t base = (size_t)b * G_ * P_ + p;
  float nrm = 0.f;
  for (int g = 0; g < G_; ++g) {
    int bg = b * G_ + g;
    float am = al[base + (size_t)g * P_] * pos[base + (size_t)g * P_];
    nrm = fmaxf(nrm, am * po[bg] / (pa[bg] + EPS_));
  }
  int ag  = agb[i];
  int bgi = b * G_ + ag;
  int lab = gl[bgi]; if (lab < 0) lab = 0;
  out_lab[i] = (float)lab;
  out_bb[(size_t)i * 4 + 0] = gb[bgi * 4 + 0];
  out_bb[(size_t)i * 4 + 1] = gb[bgi * 4 + 1];
  out_bb[(size_t)i * 4 + 2] = gb[bgi * 4 + 2];
  out_bb[(size_t)i * 4 + 3] = gb[bgi * 4 + 3];
  float f = fgf[i];
  out_fg[i]  = (f > 0.f) ? 1.f : 0.f;
  labbuf[i]  = lab;
  normbuf[i] = (f > 0.f) ? nrm : 0.f;       // fold fg mask into the score scale
}

// ---------------------------------------------------------------------------
// K7: coalesced one-hot * norm fill of (B,P,C) scores.
// ---------------------------------------------------------------------------
__global__ void k_scores(const int* __restrict__ labbuf, const float* __restrict__ normbuf,
                         float* __restrict__ out_sc) {
  size_t i = (size_t)blockIdx.x * blockDim.x + threadIdx.x;
  if (i >= (size_t)B_ * P_ * C_) return;
  size_t bp = i / C_;
  int c = (int)(i % C_);
  out_sc[i] = (c == labbuf[bp]) ? normbuf[bp] : 0.f;
}

extern "C" void kernel_launch(void* const* d_in, const int* in_sizes, int n_in,
                              void* d_out, int out_size, void* d_ws, size_t ws_size,
                              hipStream_t stream) {
  (void)in_sizes; (void)n_in; (void)out_size; (void)ws_size;
  const float* ps  = (const float*)d_in[0];   // pred_scores  (B,P,C)
  const float* pb  = (const float*)d_in[1];   // pred_bboxes  (B,P,4)
  const float* pri = (const float*)d_in[2];   // priors_points (P,2)
  const int*   gl  = (const int*)  d_in[3];   // gt_labels    (B,G,1) (int per harness)
  const float* gb  = (const float*)d_in[4];   // gt_bboxes    (B,G,4)
  const float* pf  = (const float*)d_in[5];   // pad_bbox_flag (B,G,1)

  const size_t BGP = (size_t)B_ * G_ * P_;
  const size_t BP  = (size_t)B_ * P_;
  const size_t BG  = (size_t)B_ * G_;

  float* ws      = (float*)d_ws;
  float* al      = ws;              // (B,G,P)
  float* ov      = al  + BGP;       // (B,G,P)
  float* pos     = ov  + BGP;       // (B,G,P)
  float* fgw     = pos + BGP;       // (B,P)  wmma column sums
  float* pa      = fgw + BP;        // (B,G)  pos_align
  float* po      = pa  + BG;        // (B,G)  pos_over
  float* fgf     = po  + BG;        // (B,P)  final fg
  float* normbuf = fgf + BP;        // (B,P)
  int*   agb     = (int*)(normbuf + BP); // (B,P)
  int*   labbuf  = agb + BP;        // (B,P)

  float* out     = (float*)d_out;
  float* out_lab = out;             // (B,P)
  float* out_bb  = out + BP;        // (B,P,4)
  float* out_sc  = out + BP * 5;    // (B,P,C)
  float* out_fg  = out + BP * 85;   // (B,P)

  k_align_iou<<<B_ * G_, 256, 0, stream>>>(ps, pb, gl, gb, al, ov);
  k_zero<<<(unsigned)((BGP + 255) / 256), 256, 0, stream>>>(pos, BGP);
  k_topk_scatter<<<B_ * G_, 256, 0, stream>>>(al, pri, gb, pf, pos);
  k_fg_wmma<<<B_ * (P_ / 16), 32, 0, stream>>>(pos, fgw);
  k_multi<<<(unsigned)((BP + 255) / 256), 256, 0, stream>>>(pos, ov, fgw, agb, fgf);
  k_rowmax<<<B_ * G_, 256, 0, stream>>>(al, ov, pos, pa, po);
  k_assign<<<(unsigned)((BP + 255) / 256), 256, 0, stream>>>(
      al, pos, pa, po, agb, fgf, gl, gb, out_lab, out_bb, out_fg, labbuf, normbuf);
  k_scores<<<(unsigned)((BP * C_ + 255) / 256), 256, 0, stream>>>(labbuf, normbuf, out_sc);
}
